// AlexNetOWT_BN_54563264528538
// MI455X (gfx1250) — matrix operand order repaired and checked by reading, not compiled, not run on hardware
//
#include <hip/hip_runtime.h>

// ---------------------------------------------------------------------------
// Fully-fused binarized CNN forward for MI455X (gfx1250, wave32).
// One wave == one image; all intermediates in LDS (f16).
//
// Conv mapping (implicit GEMM, multi-output-position packing):
//   N = COUT * (QY*QX block positions) = 16, K = CIN * (2+QY)*(2+QX),
//   M = 16 output blocks -> one v_wmma_f32_16x16x32_f16 = up to 64 outputs.
// Non-pool layers fuse BN+ReLU+binarize into the WMMA epilogue (no sums
// round-trip). Pool layers store block-ordered sums (2x2 block == pool
// window for conv2). Only padded-buffer *borders* are zeroed.
// ---------------------------------------------------------------------------

typedef __attribute__((ext_vector_type(16))) _Float16 v16h;
typedef __attribute__((ext_vector_type(8)))  float    v8f;

#define WPB 4  // waves (= images) per 128-thread block

struct Params {
  const float* p[33];
  float* out;
};

__device__ __forceinline__ void lds_fence() {
  asm volatile("s_wait_dscnt 0" ::: "memory");
}

__device__ __forceinline__ float fsign(float x) {
  return (x > 0.f) ? 1.f : ((x < 0.f) ? -1.f : 0.f);
}

// Zero only the 1-wide border of a padded [C][(OH+2)][(OW+2)] map.
template<int C, int OH, int OW>
__device__ void zero_border(_Float16* __restrict__ actp, int lane) {
  constexpr int PW = OW + 2, PHW = (OH + 2) * PW;
  constexpr int PER = 2 * PW + 2 * OH;
  for (int i = lane; i < C * PER; i += 32) {
    const int c = i / PER, r = i % PER;
    int idx;
    if (r < 2 * PW)
      idx = (r < PW) ? r : ((OH + 1) * PW + r - PW);       // top/bottom rows
    else {
      const int rr = r - 2 * PW;                            // side columns
      idx = (1 + (rr >> 1)) * PW + ((rr & 1) ? (OW + 1) : 0);
    }
    actp[c * PHW + idx] = (_Float16)0.f;
  }
}

// ---------------------------------------------------------------------------
// Binarized 3x3 conv (stride 1, pad 1), QY x QX outputs per WMMA column.
// actp: [CIN][(H+2)][(W+2)] f16, zero border.
// FUSE=false: dst = cnv[16][NB] raw sums, n = o*NQ + dy*QX + dx, block-major.
// FUSE=true : dst = next padded act [COUT][(H+2)][(W+2)], BN+ReLU+binarize
//             applied in the epilogue (bnb/bng/bnbe/bnm/bnv params).
// ---------------------------------------------------------------------------
template<int CIN, int COUT, int H, int W, int QY, int QX, bool FUSE>
__device__ void binconv(const _Float16* __restrict__ actp,
                        _Float16* __restrict__ dst,
                        const float* __restrict__ w, int lane,
                        const float* bnb = nullptr, const float* bng = nullptr,
                        const float* bnbe = nullptr, const float* bnm = nullptr,
                        const float* bnv = nullptr) {
  constexpr int NQ  = QY * QX;
  static_assert(COUT * NQ == 16, "N packing must fill 16 columns");
  constexpr int WH  = 2 + QY, WW = 2 + QX, WSZ = WH * WW;
  constexpr int K   = CIN * WSZ;
  constexpr int NK  = (K + 31) / 32;
  constexpr int BH  = H / QY, BW = W / QX, NB = BH * BW;
  constexpr int NT  = (NB + 15) / 16;
  constexpr int PW  = W + 2;
  constexpr int PHW = (H + 2) * PW;
  const int lhi = lane >> 4;
  const int lm  = lane & 15;

  // Column n = (o, dy, dx).
  const int o  = lm / NQ;
  const int dq = lm % NQ;
  const int dy = dq / QX, dx = dq % QX;

  // ---- B fragments (K x 16): k = i + 16*lhi per 32-chunk; col n = lm.
  // Weight w[o][ci][wy-dy][wx-dx] at window position (wy,wx); 0 elsewhere.
  v16h bf[NK];
  #pragma unroll
  for (int c = 0; c < NK; ++c) {
    #pragma unroll
    for (int i = 0; i < 16; ++i) {
      const int k  = c * 32 + i + lhi * 16;
      const bool kv = (k < K);
      const int kk = kv ? k : 0;
      const int ci = kk / WSZ, r = kk % WSZ;
      const int ky = r / WW - dy, kx = r % WW - dx;
      const bool inker = (ky >= 0) & (ky < 3) & (kx >= 0) & (kx < 3);
      const float wv = w[o * (CIN * 9) + ci * 9 + (inker ? (ky * 3 + kx) : 0)];
      bf[c][i] = (_Float16)((kv & inker) ? fsign(wv) : 0.f);
    }
  }

  // ---- A LDS offsets (constant across tiles); k>=K clamps to 0 (B zero).
  int aoff[NK][16];
  #pragma unroll
  for (int c = 0; c < NK; ++c) {
    #pragma unroll
    for (int i = 0; i < 16; ++i) {
      const int vv = i >> 1, hh = i & 1;
      int k = c * 32 + lhi * 16 + i;
      if (k >= K) k = 0;
      const int ci = k / WSZ, r = k % WSZ;
      aoff[c][i] = ci * PHW + (r / WW) * PW + (r % WW);
    }
  }

  // ---- Per-lane fused BN constants (hoisted; o is fixed per lane).
  float sc = 0.f, t1 = 0.f, t2 = 0.f;
  if constexpr (FUSE) {
    sc = bng[o] * rsqrtf(bnv[o] + 1e-5f);
    t1 = bnb[o] - bnm[o];
    t2 = bnbe[o];
  }

  for (int t = 0; t < NT; ++t) {
    int blk = t * 16 + lm;
    if (blk >= NB) blk = NB - 1;                 // clamp; masked at store
    const int sbase = (blk / BW) * (QY * PW) + (blk % BW) * QX;
    v8f acc = {};
    #pragma unroll
    for (int c = 0; c < NK; ++c) {
      v16h af;
      #pragma unroll
      for (int i = 0; i < 16; ++i) af[i] = actp[sbase + aoff[c][i]];
      acc = __builtin_amdgcn_wmma_f32_16x16x32_f16(
          false, af, false, bf[c], (short)0, acc, false, false);
    }
    // D: lane holds col n=lm, rows m = lhi*8 + j -> 8 consecutive blocks.
    #pragma unroll
    for (int j = 0; j < 8; ++j) {
      const int bb = t * 16 + lhi * 8 + j;
      if (bb < NB) {
        if constexpr (FUSE) {
          const int by = bb / BW, bx = bb % BW;
          const int oy = QY * by + dy, ox = QX * bx + dx;
          const float z = sc * (acc[j] + t1) + t2;
          dst[o * PHW + (oy + 1) * PW + (ox + 1)] =
              (_Float16)((z > 0.f) ? 1.f : 0.f);   // sign(relu(bn(conv+b)))
        } else {
          dst[lm * NB + bb] = (_Float16)acc[j];
        }
      }
    }
  }
}

// ---------------------------------------------------------------------------
// 2x2 maxpool -> BN -> ReLU -> binarize (pool layers only).
// Reads block-ordered sums from binconv<...,QY,QX,false>.
// ---------------------------------------------------------------------------
template<int C, int H, int W, int QY, int QX>
__device__ void poolact(const _Float16* __restrict__ cnv,
                        _Float16* __restrict__ actp, float* __restrict__ scr,
                        int lane,
                        const float* __restrict__ b, const float* __restrict__ g,
                        const float* __restrict__ be, const float* __restrict__ m,
                        const float* __restrict__ v) {
  constexpr int NQ  = QY * QX;
  constexpr int BW  = W / QX, NB = (H / QY) * BW;
  constexpr int OH  = H / 2, OW = W / 2;
  constexpr int PW  = OW + 2, PHW = (OH + 2) * PW;
  constexpr int N   = C * OH * OW;

  if (lane < C) {
    scr[lane]      = g[lane] * rsqrtf(v[lane] + 1e-5f);
    scr[16 + lane] = b[lane] - m[lane];
    scr[32 + lane] = be[lane];
  }
  zero_border<C, OH, OW>(actp, lane);
  lds_fence();

  for (int idx = lane; idx < N; idx += 32) {
    const int c = idx / (OH * OW), r = idx % (OH * OW);
    const int py = r / OW, px = r % OW;
    float val = -3.0e38f;
    #pragma unroll
    for (int sy = 0; sy < 2; ++sy)
      #pragma unroll
      for (int sx = 0; sx < 2; ++sx) {
        const int oy = 2 * py + sy, ox = 2 * px + sx;
        const int n   = c * NQ + (oy % QY) * QX + (ox % QX);
        const int blk = (oy / QY) * BW + (ox / QX);
        val = fmaxf(val, (float)cnv[n * NB + blk]);
      }
    const float z = scr[c] * (val + scr[16 + c]) + scr[32 + c];
    actp[c * PHW + (py + 1) * PW + (px + 1)] =
        (_Float16)((z > 0.f) ? 1.f : 0.f);
  }
}

// ---------------------------------------------------------------------------
__global__ __launch_bounds__(32 * WPB, 1) void bnn_fused(Params P) {
  __shared__ _Float16 s_bufA[WPB][3600];  // padded maps / block sums
  __shared__ _Float16 s_bufB[WPB][3600];
  __shared__ float    s_scr[WPB][64];

  const int lane = threadIdx.x & 31;
  const int wv   = threadIdx.x >> 5;
  const int img  = blockIdx.x * WPB + wv;  // grid sized exactly: always valid

  _Float16* A = s_bufA[wv];
  _Float16* B = s_bufB[wv];
  float*    scr = s_scr[wv];

  // ---- stage 0: binarized input into padded 1x30x30 (A) ----
  {
    const float* x = P.p[0] + (size_t)img * 784;
    zero_border<1, 28, 28>(A, lane);
    for (int i = lane; i < 784; i += 32)
      A[(i / 28 + 1) * 30 + (i % 28 + 1)] = (_Float16)fsign(x[i]);
    lds_fence();
  }

  // ---- conv1 (1->4, 28x28, 2x2 blocks) + bn1 + relu fused -> B ----
  zero_border<4, 28, 28>(B, lane);
  lds_fence();
  binconv<1, 4, 28, 28, 2, 2, true>(A, B, P.p[1], lane,
                                    P.p[2], P.p[3], P.p[4], P.p[5], P.p[6]);
  lds_fence();

  // ---- conv2 (4->4, 28x28, 2x2 blocks == pool windows): sums -> A ----
  binconv<4, 4, 28, 28, 2, 2, false>(B, A, P.p[7], lane);
  lds_fence();
  poolact<4, 28, 28, 2, 2>(A, B, scr, lane,
                           P.p[8], P.p[9], P.p[10], P.p[11], P.p[12]);
  lds_fence();

  // ---- conv3 (4->8, 14x14, 1x2 blocks) + bn3 + relu fused -> A ----
  zero_border<8, 14, 14>(A, lane);
  lds_fence();
  binconv<4, 8, 14, 14, 1, 2, true>(B, A, P.p[13], lane,
                                    P.p[14], P.p[15], P.p[16], P.p[17], P.p[18]);
  lds_fence();

  // ---- conv4 (8->8, 14x14, 1x2 blocks): sums -> B, pool -> A ----
  binconv<8, 8, 14, 14, 1, 2, false>(A, B, P.p[19], lane);
  lds_fence();
  poolact<8, 14, 14, 1, 2>(B, A, scr, lane,
                           P.p[20], P.p[21], P.p[22], P.p[23], P.p[24]);
  lds_fence();

  // ---- conv5 (8->16, 7x7, 1x1): raw sums -> B[16][49] ----
  binconv<8, 16, 7, 7, 1, 1, false>(A, B, P.p[25], lane);
  lds_fence();

  // ---- head: global-avg-pool + bias5 -> bn5 -> relu -> binarize ----
  if (lane < 16) {
    float s = 0.f;
    #pragma unroll
    for (int i = 0; i < 49; ++i) s += (float)B[lane * 49 + i];
    const float mean = s * (1.f / 49.f) + P.p[26][lane];           // + b5
    const float sc = P.p[27][lane] * rsqrtf(P.p[30][lane] + 1e-5f);
    const float z  = sc * (mean - P.p[29][lane]) + P.p[28][lane];
    scr[lane] = (z > 0.f) ? 1.f : 0.f;                             // sign(relu)
  }
  lds_fence();

  // ---- binarized linear 16->10 ----
  if (lane < 10) {
    const float* wl = P.p[31];
    float logit = P.p[32][lane];                                    // bl
    #pragma unroll
    for (int c = 0; c < 16; ++c)
      logit += scr[c] * fsign(wl[lane * 16 + c]);
    scr[16 + lane] = logit;
  }
  lds_fence();

  // ---- log_softmax over 10 ----
  if (lane < 10) {
    float mx = -1e30f;
    #pragma unroll
    for (int j = 0; j < 10; ++j) mx = fmaxf(mx, scr[16 + j]);
    float se = 0.f;
    #pragma unroll
    for (int j = 0; j < 10; ++j) se += __expf(scr[16 + j] - mx);
    P.out[(size_t)img * 10 + lane] = scr[16 + lane] - mx - __logf(se);
  }
}

// ---------------------------------------------------------------------------
extern "C" void kernel_launch(void* const* d_in, const int* in_sizes, int n_in,
                              void* d_out, int out_size, void* d_ws, size_t ws_size,
                              hipStream_t stream) {
  (void)in_sizes; (void)n_in; (void)out_size; (void)d_ws; (void)ws_size;
  Params P;
  for (int i = 0; i < 33; ++i) P.p[i] = (const float*)d_in[i];
  P.out = (float*)d_out;

  const int images = 8192;
  const int blocks = images / WPB;  // 2048 blocks x 128 threads (4 waves)
  bnn_fused<<<blocks, 32 * WPB, 0, stream>>>(P);
}
